// MambaIntegerBlock_7249904795695
// MI455X (gfx1250) — matrix operand order repaired and checked by reading, not compiled, not run on hardware
//
#include <hip/hip_runtime.h>
#include <hip/hip_bf16.h>

// ---------------------------------------------------------------------------
// Types for CDNA5 WMMA
// ---------------------------------------------------------------------------
typedef __bf16 bf16;
typedef __attribute__((ext_vector_type(16))) __bf16 v16bf;
typedef __attribute__((ext_vector_type(8)))  __bf16 v8bf;
typedef __attribute__((ext_vector_type(8)))  float  v8f;

static __device__ __forceinline__ bf16 f2bf(float f) {
    unsigned u = __builtin_bit_cast(unsigned, f);
    unsigned r = u + 0x7FFFu + ((u >> 16) & 1u);   // round-to-nearest-even
    unsigned short s = (unsigned short)(r >> 16);
    return __builtin_bit_cast(bf16, s);
}

// Problem constants
#define BSZ      2
#define SEQ      1024
#define DMODEL   1024
#define DINNER   2048
#define DTRANK   64
#define DSTATE   16
#define NTOK     (BSZ * SEQ)             // 2048
#define XDBL_N   (DTRANK + 2 * DSTATE)   // 96

// ---------------------------------------------------------------------------
// f32 -> bf16 elementwise conversion (weights)
// ---------------------------------------------------------------------------
__global__ __launch_bounds__(256) void cvt_f32_bf16(const float* __restrict__ in,
                                                    bf16* __restrict__ out, int n) {
    int i = blockIdx.x * 256 + threadIdx.x;
    if (i < n) out[i] = f2bf(in[i]);
}

// ---------------------------------------------------------------------------
// bitshift_norm: per-token variance over DMODEL, k = clip(floor(0.5*log2(var+eps)),0)
// h = x * 2^-k * gamma  -> bf16.  One block per token.
// ---------------------------------------------------------------------------
__global__ __launch_bounds__(256) void norm_kernel(const float* __restrict__ x,
                                                   const float* __restrict__ gamma,
                                                   bf16* __restrict__ h) {
    __shared__ float red[256];
    const int t = blockIdx.x;          // token 0..NTOK-1
    const int tid = threadIdx.x;
    const float* xr = x + (size_t)t * DMODEL;

    float s = 0.f;
    for (int i = tid; i < DMODEL; i += 256) { float v = xr[i]; s += v * v; }
    red[tid] = s;
    __syncthreads();
    for (int off = 128; off; off >>= 1) {
        if (tid < off) red[tid] += red[tid + off];
        __syncthreads();
    }
    float var = red[0] * (1.0f / DMODEL);
    float kf = floorf(0.5f * log2f(var + 1e-9f));
    if (kf < 0.f) kf = 0.f;
    float scl = exp2f(-kf);

    bf16* hr = h + (size_t)t * DMODEL;
    for (int i = tid; i < DMODEL; i += 256)
        hr[i] = f2bf(xr[i] * scl * gamma[i]);
}

// ---------------------------------------------------------------------------
// Generic WMMA bf16 GEMM:  out[M,N] = A[M,K] @ W[N,K]^T (+ addend)
// A, W row-major bf16. K % 32 == 0, M % 256 == 0, N % 16 == 0.
// Block: 256 threads = 8 waves (4M x 2N); wave tile 64x64 (16 WMMA/k-step);
// block tile 256x128. Branchless inner loop: OOB columns are clamped (their
// tiles accumulate garbage but are never stored).
// Index-form addressing (not pointer arrays) so the compiler keeps global
// provenance and emits GLOBAL_LOAD/GLOBAL_PREFETCH, not FLAT (no DScnt tie-up).
// Fragment layouts follow CDNA5 ISA 7.12.2 (wave32).
// ---------------------------------------------------------------------------
__global__ __launch_bounds__(256) void gemm_bf16_wmma(const bf16* __restrict__ A,
                                                      const bf16* __restrict__ W,
                                                      float* __restrict__ out,
                                                      const float* __restrict__ addend,
                                                      int M, int N, int K) {
    const int lane  = threadIdx.x & 31;
    const int wave  = threadIdx.x >> 5;        // 0..7
    const int wm    = wave & 3;                // 4 waves along M
    const int wn    = wave >> 2;               // 2 waves along N
    const int mBase = blockIdx.x * 256 + wm * 64;
    const int nBase = blockIdx.y * 128 + wn * 64;
    const int l15   = lane & 15;
    const int hi    = lane >> 4;               // lane half

    // Clamped B columns (branchless OOB safety; OOB tiles never stored)
    int bcol[4];
#pragma unroll
    for (int nt = 0; nt < 4; ++nt) {
        int col = nBase + nt * 16 + l15;
        bcol[nt] = col < N ? col : N - 1;
    }

    v8f acc[4][4] = {};

    for (int k = 0; k < K; k += 32) {
        // ---- A fragments (16x32 bf16, lane=row, halves split K) ----
        v16bf a[4];
#pragma unroll
        for (int mt = 0; mt < 4; ++mt) {
            const bf16* p = A + (size_t)(mBase + mt * 16 + l15) * K + k + hi * 8;
            v8bf c0 = *(const v8bf*)p;          // K = k+hi*8    .. +7
            v8bf c1 = *(const v8bf*)(p + 16);   // K = k+16+hi*8 .. +7
#pragma unroll
            for (int i = 0; i < 8; ++i) { a[mt][i] = c0[i]; a[mt][8 + i] = c1[i]; }
            __builtin_prefetch(p + 64, 0, 1);   // one 128B line (2 k-steps) ahead
        }
        // ---- B fragments (32x16 bf16, lane=col, halves split K) ----
        v16bf b[4];
#pragma unroll
        for (int nt = 0; nt < 4; ++nt) {
            const bf16* p = W + (size_t)bcol[nt] * K + k + hi * 8;
            v8bf c0 = *(const v8bf*)p;
            v8bf c1 = *(const v8bf*)(p + 16);
#pragma unroll
            for (int i = 0; i < 8; ++i) { b[nt][i] = c0[i]; b[nt][8 + i] = c1[i]; }
            __builtin_prefetch(p + 64, 0, 1);
        }
        // ---- 16 WMMAs ----
#pragma unroll
        for (int mt = 0; mt < 4; ++mt)
#pragma unroll
            for (int nt = 0; nt < 4; ++nt)
                acc[mt][nt] = __builtin_amdgcn_wmma_f32_16x16x32_bf16(
                    false, a[mt], false, b[nt], (short)0, acc[mt][nt], false, false);
    }

    // ---- epilogue: C/D layout lane=col, VGPR r = row (hi*8 + r) ----
#pragma unroll
    for (int mt = 0; mt < 4; ++mt)
#pragma unroll
        for (int nt = 0; nt < 4; ++nt) {
            const int col = nBase + nt * 16 + l15;
            if (col < N) {   // uniform per 16-wide tile (N % 16 == 0)
#pragma unroll
                for (int r = 0; r < 8; ++r) {
                    const int row = mBase + mt * 16 + hi * 8 + r;
                    const size_t o = (size_t)row * N + col;
                    float v = acc[mt][nt][r];
                    if (addend) v += addend[o];
                    out[o] = v;
                }
            }
        }
}

// ---------------------------------------------------------------------------
// Causal depthwise conv (width 4, left pad 3) + bias + square.
// xz layout [NTOK, 2*DINNER]; x half is cols [0, DINNER).
// Writes xs_f32 and xs_bf16, both [NTOK, DINNER].
// ---------------------------------------------------------------------------
__global__ __launch_bounds__(256) void conv_sq_kernel(const float* __restrict__ xz,
                                                      const float* __restrict__ conv_w,
                                                      const float* __restrict__ conv_b,
                                                      float* __restrict__ xs_f32,
                                                      bf16* __restrict__ xs_bf16) {
    const int idx = blockIdx.x * 256 + threadIdx.x;   // over NTOK*DINNER
    const int d = idx & (DINNER - 1);
    const int t = idx >> 11;                          // DINNER == 2048
    const int b = t >> 10;                            // SEQ == 1024
    const int l = t & (SEQ - 1);

    float acc = conv_b[d];
#pragma unroll
    for (int j = 0; j < 4; ++j) {
        const int ll = l - 3 + j;
        if (ll >= 0)
            acc += conv_w[d * 4 + j] * xz[((size_t)(b * SEQ + ll)) * (2 * DINNER) + d];
    }
    const float sq = acc * acc;
    xs_f32[(size_t)t * DINNER + d]  = sq;
    xs_bf16[(size_t)t * DINNER + d] = f2bf(sq);
}

// ---------------------------------------------------------------------------
// Slice dt (first DTRANK cols of x_dbl) -> bf16
// ---------------------------------------------------------------------------
__global__ __launch_bounds__(256) void dt_slice_kernel(const float* __restrict__ x_dbl,
                                                       bf16* __restrict__ dt_bf16) {
    const int idx = blockIdx.x * 256 + threadIdx.x;   // over NTOK*DTRANK
    const int j = idx & (DTRANK - 1);
    const int t = idx >> 6;
    dt_bf16[idx] = f2bf(x_dbl[(size_t)t * XDBL_N + j]);
}

// ---------------------------------------------------------------------------
// Fused selective scan. One thread per (batch, d, n):
//   decay = floor(clip(base+mod,0,32767)) * 2^-shift
//   h     = decay*h + x*B ;  y[d] = sum_n h*C[n]  (16-lane shfl_xor reduce)
//   y *= z*z ; write bf16 for the output GEMM. hs never materialized.
// Block = 256 thr = 16 d-channels x 16 states; 2 d per wave.
// unroll 4 so the compiler batches independent loads ahead of the recurrence.
// ---------------------------------------------------------------------------
__global__ __launch_bounds__(256) void scan_kernel(const float* __restrict__ xs,
                                                   const float* __restrict__ x_dbl,
                                                   const float* __restrict__ decay_mod,
                                                   const float* __restrict__ base_nums,
                                                   const int*   __restrict__ shifts,
                                                   const float* __restrict__ xz,
                                                   bf16* __restrict__ y_bf16) {
    const int tid   = threadIdx.x;
    const int n     = tid & 15;
    const int dl    = tid >> 4;                    // 0..15
    const int batch = blockIdx.x >> 7;             // DINNER/16 = 128 blocks/batch
    const int dblk  = blockIdx.x & 127;
    const int d     = dblk * 16 + dl;

    const float base  = base_nums[d * DSTATE + n];
    const float scale = exp2f(-(float)shifts[d * DSTATE + n]);
    float h = 0.f;
    const int t0 = batch * SEQ;

#pragma unroll 4
    for (int l = 0; l < SEQ; ++l) {
        const int t = t0 + l;
        const float dm = decay_mod[(size_t)t * DINNER + d];
        const float xv = xs[(size_t)t * DINNER + d];
        const float Bv = x_dbl[(size_t)t * XDBL_N + DTRANK + n];
        const float Cv = x_dbl[(size_t)t * XDBL_N + DTRANK + DSTATE + n];

        float dn = floorf(fminf(fmaxf(base + dm, 0.f), 32767.f));
        h = dn * scale * h + xv * Bv;

        float c = h * Cv;
        c += __shfl_xor(c, 1, 16);
        c += __shfl_xor(c, 2, 16);
        c += __shfl_xor(c, 4, 16);
        c += __shfl_xor(c, 8, 16);

        if (n == 0) {
            const float zv = xz[(size_t)t * (2 * DINNER) + DINNER + d];
            y_bf16[(size_t)t * DINNER + d] = f2bf(c * zv * zv);
        }
    }
}

// ---------------------------------------------------------------------------
// Host orchestration
// ---------------------------------------------------------------------------
static inline size_t alignup(size_t x) { return (x + 255) & ~(size_t)255; }

extern "C" void kernel_launch(void* const* d_in, const int* in_sizes, int n_in,
                              void* d_out, int out_size, void* d_ws, size_t ws_size,
                              hipStream_t stream) {
    (void)in_sizes; (void)n_in; (void)out_size; (void)ws_size;

    const float* hidden   = (const float*)d_in[0];   // [2,1024,1024]
    const float* gamma    = (const float*)d_in[1];   // [1024]
    const float* w_in     = (const float*)d_in[2];   // [4096,1024]
    const float* conv_w   = (const float*)d_in[3];   // [2048,1,4]
    const float* conv_b   = (const float*)d_in[4];   // [2048]
    const float* w_x      = (const float*)d_in[5];   // [96,2048]
    const float* w_dt     = (const float*)d_in[6];   // [2048,64]
    const float* w_out    = (const float*)d_in[7];   // [1024,2048]
    const float* base_nm  = (const float*)d_in[8];   // [2048,16]
    const int*   shifts   = (const int*)  d_in[9];   // [2048,16]
    float*       out      = (float*)d_out;           // [2,1024,1024]

    // Workspace carve-out
    char* ws = (char*)d_ws;
    size_t off = 0;
    auto take = [&](size_t bytes) { char* p = ws + off; off += alignup(bytes); return p; };

    bf16*  w_in_bf  = (bf16*) take((size_t)2 * DINNER * DMODEL * sizeof(bf16));
    bf16*  w_x_bf   = (bf16*) take((size_t)XDBL_N * DINNER * sizeof(bf16));
    bf16*  w_dt_bf  = (bf16*) take((size_t)DINNER * DTRANK * sizeof(bf16));
    bf16*  w_out_bf = (bf16*) take((size_t)DMODEL * DINNER * sizeof(bf16));
    bf16*  h_bf     = (bf16*) take((size_t)NTOK * DMODEL * sizeof(bf16));
    float* xz       = (float*)take((size_t)NTOK * 2 * DINNER * sizeof(float));
    float* xs_f32   = (float*)take((size_t)NTOK * DINNER * sizeof(float));
    bf16*  xs_bf    = (bf16*) take((size_t)NTOK * DINNER * sizeof(bf16));
    float* x_dbl    = (float*)take((size_t)NTOK * XDBL_N * sizeof(float));
    bf16*  dt_bf    = (bf16*) take((size_t)NTOK * DTRANK * sizeof(bf16));
    float* dec_mod  = (float*)take((size_t)NTOK * DINNER * sizeof(float));
    bf16*  y_bf     = (bf16*) take((size_t)NTOK * DINNER * sizeof(bf16));

    // 1) weight conversions to bf16
    cvt_f32_bf16<<<(2 * DINNER * DMODEL) / 256, 256, 0, stream>>>(w_in,  w_in_bf,  2 * DINNER * DMODEL);
    cvt_f32_bf16<<<(XDBL_N * DINNER) / 256, 256, 0, stream>>>(w_x,   w_x_bf,   XDBL_N * DINNER);
    cvt_f32_bf16<<<(DINNER * DTRANK) / 256, 256, 0, stream>>>(w_dt,  w_dt_bf,  DINNER * DTRANK);
    cvt_f32_bf16<<<(DMODEL * DINNER) / 256, 256, 0, stream>>>(w_out, w_out_bf, DMODEL * DINNER);

    // 2) bitshift norm -> h (bf16)
    norm_kernel<<<NTOK, 256, 0, stream>>>(hidden, gamma, h_bf);

    // 3) GEMM1: xz[NTOK, 4096] = h @ w_in^T
    gemm_bf16_wmma<<<dim3(NTOK / 256, (2 * DINNER) / 128), 256, 0, stream>>>(
        h_bf, w_in_bf, xz, nullptr, NTOK, 2 * DINNER, DMODEL);

    // 4) causal conv4 + square
    conv_sq_kernel<<<(NTOK * DINNER) / 256, 256, 0, stream>>>(xz, conv_w, conv_b, xs_f32, xs_bf);

    // 5) GEMM2: x_dbl[NTOK, 96] = xs @ w_x^T   (grid.y rounds 96 -> 128, guarded)
    gemm_bf16_wmma<<<dim3(NTOK / 256, (XDBL_N + 127) / 128), 256, 0, stream>>>(
        xs_bf, w_x_bf, x_dbl, nullptr, NTOK, XDBL_N, DINNER);

    // 6) dt slice -> bf16
    dt_slice_kernel<<<(NTOK * DTRANK) / 256, 256, 0, stream>>>(x_dbl, dt_bf);

    // 7) GEMM3: decay_mod[NTOK, 2048] = dt @ w_dt^T
    gemm_bf16_wmma<<<dim3(NTOK / 256, DINNER / 128), 256, 0, stream>>>(
        dt_bf, w_dt_bf, dec_mod, nullptr, NTOK, DINNER, DTRANK);

    // 8) fused selective scan -> y * z^2 (bf16)
    scan_kernel<<<BSZ * (DINNER / 16), 256, 0, stream>>>(
        xs_f32, x_dbl, dec_mod, base_nm, shifts, xz, y_bf);

    // 9) GEMM4: out = y @ w_out^T + residual
    gemm_bf16_wmma<<<dim3(NTOK / 256, DMODEL / 128), 256, 0, stream>>>(
        y_bf, w_out_bf, out, hidden, NTOK, DMODEL, DINNER);
}
